// RNN_63385127354782
// MI455X (gfx1250) — matrix operand order repaired and checked by reading, not compiled, not run on hardware
//
#include <hip/hip_runtime.h>
#include <hip/hip_bf16.h>

typedef _Float16 h16;
typedef __attribute__((ext_vector_type(16))) _Float16 v16h;
typedef __attribute__((ext_vector_type(8)))  float    v8f;

#define RNN_T 30
#define RNN_I 3
#define RNN_H 32
#define RNN_L 5

// fused-kernel config
#define FWAVES 2
#define TRAJ_HALVES (16 * 960)
#define ACT1_HALVES (16 * 512)
#define HX_HALVES   (16 * 32)
#define WAVE_HALVES (TRAJ_HALVES + ACT1_HALVES + HX_HALVES)

// split-kernel config
#define RWAVES 8                    // rnn kernel waves/block
#define MWAVES 4                    // mlp kernel waves/block
#define MACT_HALVES (16 * 512 + 16 * 256)   // per-wave act region (24KB)

// MLP weight sizes (elements)
#define N1SZ (512 * 960)
#define N2SZ (256 * 512)
#define N3SZ (128 * 256)
#define N4SZ (64 * 128)
#define WTOT (N1SZ + N2SZ + N3SZ + N4SZ)

// ---------------- fast branchless tanh: v_exp_f32 + v_rcp_f32 ----------------
__device__ __forceinline__ float fast_tanh(float x) {
  float xc = fminf(fmaxf(x, -10.0f), 10.0f);
  float t  = __builtin_amdgcn_exp2f(xc * 2.8853900817779268f);  // exp(2x)
  float r  = __builtin_amdgcn_rcpf(t + 1.0f);
  return (t - 1.0f) * r;
}

// ---------------- fragment loaders (CDNA5 ISA 7.12.2, 16-bit layouts) ----------------
// A 16x32 (MxK): lane<16 -> row=lane, K 0..7 & 16..23 ; lane>=16 -> row=lane-16, K 8..15 & 24..31
// => two contiguous 16B runs per lane. Works for LDS or global h16 tiles (generic pointer).
__device__ __forceinline__ v16h afrag_h16(const h16* tile, int stride, int colbase, int lane) {
  const h16* p = tile + (lane & 15) * stride + colbase + ((lane >= 16) ? 8 : 0);
  union { uint4 q[2]; v16h v; } cv;
  cv.q[0] = *(const uint4*)p;
  cv.q[1] = *(const uint4*)(p + 16);
  return cv.v;
}

// A fragment from global f32 rows (initial h0)
__device__ __forceinline__ v16h afrag_gf32(const float* M, int stride, int row, int lane) {
  v16h r;
  const float* rp = M + (size_t)row * stride;
#pragma unroll
  for (int j = 0; j < 8; ++j) {
    int k = ((j >= 4) ? 16 : 0) + ((lane >= 16) ? 8 : 0) + 2 * (j & 3);
    float2 f = *(const float2*)(rp + k);
    r[2 * j] = (h16)f.x; r[2 * j + 1] = (h16)f.y;
  }
  return r;
}

// B 32x16 (KxN) for X@W^T, W [N x K] row-major f16: lane<16 -> K 0..7 & 16..23, lane>=16 -> K 8..15 & 24..31
__device__ __forceinline__ v16h bfrag_h16(const h16* W, int Kin, int nrow, int colbase, int lane) {
  const h16* p = W + (size_t)nrow * Kin + colbase + ((lane >= 16) ? 8 : 0);
  union { uint4 q[2]; v16h v; } cv;
  cv.q[0] = *(const uint4*)p;
  cv.q[1] = *(const uint4*)(p + 16);
  return cv.v;
}

// fallback: B fragment from global f32 with inline f32->f16 conversion
__device__ __forceinline__ v16h bfrag_gf32(const float* W, int Kin, int nrow, int colbase, int lane) {
  v16h r;
  const float* rp = W + (size_t)nrow * Kin;
#pragma unroll
  for (int j = 0; j < 8; ++j) {
    int k = colbase + ((j >= 4) ? 16 : 0) + ((lane >= 16) ? 8 : 0) + 2 * (j & 3);
    float2 f = *(const float2*)(rp + k);
    r[2 * j] = (h16)f.x; r[2 * j + 1] = (h16)f.y;
  }
  return r;
}

__device__ __forceinline__ v8f wmma_f16(v16h a, v16h b, v8f c) {
  return __builtin_amdgcn_wmma_f32_16x16x32_f16(false, a, false, b, (short)0, c, false, false);
}

// ---------------- one WMMA MLP layer ----------------
template <int KIN, int NOUT, bool RELU, bool PRE>
__device__ __forceinline__ void mlp_layer(const h16* in, const float* __restrict__ Wf,
                                          const h16* __restrict__ Wh,
                                          const float* __restrict__ bias, h16* outbuf, int lane) {
  const int mrow  = lane & 15;
  const int mbase = (lane >= 16) ? 8 : 0;
#pragma unroll 1
  for (int nt = 0; nt < NOUT / 16; ++nt) {
    v8f acc = {};
#pragma unroll 2
    for (int kt = 0; kt < KIN / 32; ++kt) {
      v16h a = afrag_h16(in, KIN, kt * 32, lane);
      v16h b = PRE ? bfrag_h16(Wh, KIN, nt * 16 + mrow, kt * 32, lane)
                   : bfrag_gf32(Wf, KIN, nt * 16 + mrow, kt * 32, lane);
      acc = wmma_f16(a, b, acc);
    }
    float bv = bias[nt * 16 + mrow];
#pragma unroll
    for (int r = 0; r < 8; ++r) {
      float v = acc[r] + bv;
      if (RELU) v = fmaxf(v, 0.0f);
      outbuf[(mbase + r) * NOUT + nt * 16 + mrow] = (h16)v;
    }
  }
  __syncthreads();
}

// ---------------- fp32 -> f16 weight pre-conversion ----------------
__global__ __launch_bounds__(256) void conv_weights_kernel(const float* __restrict__ W1,
                                                           const float* __restrict__ W2,
                                                           const float* __restrict__ W3,
                                                           const float* __restrict__ W4,
                                                           h16* __restrict__ o) {
  int i = blockIdx.x * 256 + threadIdx.x;
  if (i >= WTOT) return;
  float v;
  if (i < N1SZ) v = W1[i];
  else if (i < N1SZ + N2SZ) v = W2[i - N1SZ];
  else if (i < N1SZ + N2SZ + N3SZ) v = W3[i - N1SZ - N2SZ];
  else v = W4[i - N1SZ - N2SZ - N3SZ];
  o[i] = (h16)v;
}

// ============================================================================
// SPLIT PATH kernel 1: RNN recurrence, trajectory streamed to global scratch.
// 8 waves/block, ~29KB LDS -> many waves/SIMD to hide the serial-chain latency.
// ============================================================================
__global__ __launch_bounds__(32 * RWAVES) void rnn_traj_kernel(
    const float* __restrict__ x, const float* __restrict__ h0,
    const float* __restrict__ Wih0, const float* __restrict__ Wih,
    const float* __restrict__ Whh, const float* __restrict__ bih,
    const float* __restrict__ bhh, h16* __restrict__ trajG, int Bsz) {
  __shared__ __align__(32) h16   s_whh[RNN_L][RNN_H][RNN_H];
  __shared__ __align__(32) h16   s_wih[RNN_L - 1][RNN_H][RNN_H];
  __shared__ __align__(32) h16   s_wih0[RNN_H][RNN_H];
  __shared__ float s_bias[RNN_L][RNN_H];
  __shared__ __align__(32) h16   s_hx[RWAVES][HX_HALVES];

  const int tid   = threadIdx.x;
  const int lane  = tid & 31;
  const int wave  = tid >> 5;
  const int b0    = blockIdx.x * (RWAVES * 16) + wave * 16;
  const int mrow  = lane & 15;
  const int mbase = (lane >= 16) ? 8 : 0;

  for (int i = tid; i < RNN_L * RNN_H * RNN_H; i += blockDim.x)
    (&s_whh[0][0][0])[i] = (h16)Whh[i];
  for (int i = tid; i < (RNN_L - 1) * RNN_H * RNN_H; i += blockDim.x)
    (&s_wih[0][0][0])[i] = (h16)Wih[i];
  for (int i = tid; i < RNN_H * RNN_H; i += blockDim.x) {
    int n = i >> 5, k = i & 31;
    (&s_wih0[0][0])[i] = (k < RNN_I) ? (h16)Wih0[n * RNN_I + k] : (h16)0.0f;
  }
  for (int i = tid; i < RNN_L * RNN_H; i += blockDim.x)
    (&s_bias[0][0])[i] = bih[i] + bhh[i];
  __syncthreads();

  h16* hx = &s_hx[wave][0];

  v16h hfrag[RNN_L];
#pragma unroll
  for (int l = 0; l < RNN_L; ++l)
    hfrag[l] = afrag_gf32(h0 + (size_t)l * Bsz * RNN_H, RNN_H, b0 + mrow, lane);

#pragma unroll 1
  for (int t = 0; t < RNN_T; ++t) {
    // layer 0 (K=3 input zero-padded into the K=32 A fragment)
    v16h xa = {};
    if (lane < 16) {
      const float* xp = x + ((size_t)(b0 + lane) * RNN_T + t) * RNN_I;
      xa[0] = (h16)xp[0]; xa[1] = (h16)xp[1]; xa[2] = (h16)xp[2];
    }
    {
      float hv[2][8];
#pragma unroll
      for (int nt = 0; nt < 2; ++nt) {
        v8f acc = {};
        acc = wmma_f16(xa, bfrag_h16(&s_wih0[0][0], RNN_H, nt * 16 + mrow, 0, lane), acc);
        acc = wmma_f16(hfrag[0], bfrag_h16(&s_whh[0][0][0], RNN_H, nt * 16 + mrow, 0, lane), acc);
        float bv = s_bias[0][nt * 16 + mrow];
#pragma unroll
        for (int r = 0; r < 8; ++r) hv[nt][r] = fast_tanh(acc[r] + bv);
      }
#pragma unroll
      for (int nt = 0; nt < 2; ++nt)
#pragma unroll
        for (int r = 0; r < 8; ++r)
          hx[(mbase + r) * RNN_H + nt * 16 + mrow] = (h16)hv[nt][r];
      __syncthreads();
      hfrag[0] = afrag_h16(hx, RNN_H, 0, lane);
    }
    // layers 1..4
#pragma unroll
    for (int l = 1; l < RNN_L; ++l) {
      float hv[2][8];
#pragma unroll
      for (int nt = 0; nt < 2; ++nt) {
        v8f acc = {};
        acc = wmma_f16(hfrag[l - 1], bfrag_h16(&s_wih[l - 1][0][0], RNN_H, nt * 16 + mrow, 0, lane), acc);
        acc = wmma_f16(hfrag[l],     bfrag_h16(&s_whh[l][0][0],     RNN_H, nt * 16 + mrow, 0, lane), acc);
        float bv = s_bias[l][nt * 16 + mrow];
#pragma unroll
        for (int r = 0; r < 8; ++r) hv[nt][r] = fast_tanh(acc[r] + bv);
      }
      if (l == RNN_L - 1) {
        // stream layer-4 output to the global trajectory [B][T*32]
        // (per instruction: two 32B-contiguous runs across the lane halves)
#pragma unroll
        for (int nt = 0; nt < 2; ++nt)
#pragma unroll
          for (int r = 0; r < 8; ++r)
            trajG[(size_t)(b0 + mbase + r) * (RNN_T * RNN_H) + t * RNN_H + nt * 16 + mrow] =
                (h16)hv[nt][r];
      }
#pragma unroll
      for (int nt = 0; nt < 2; ++nt)
#pragma unroll
        for (int r = 0; r < 8; ++r)
          hx[(mbase + r) * RNN_H + nt * 16 + mrow] = (h16)hv[nt][r];
      __syncthreads();
      hfrag[l] = afrag_h16(hx, RNN_H, 0, lane);
    }
  }
}

// ============================================================================
// SPLIT PATH kernel 2: MLP head; layer-1 A fragments read straight from the
// global trajectory (global_load_b128 pairs), f16 weights from scratch.
// ============================================================================
__global__ __launch_bounds__(32 * MWAVES) void mlp_head_kernel(
    const h16* __restrict__ trajG, const h16* __restrict__ Whq,
    const float* __restrict__ b1, const float* __restrict__ b2,
    const float* __restrict__ b3, const float* __restrict__ b4,
    const float* __restrict__ W5, const float* __restrict__ b5,
    float* __restrict__ out) {
  __shared__ __align__(32) h16 s_act[MWAVES][MACT_HALVES];

  const int tid  = threadIdx.x;
  const int lane = tid & 31;
  const int wave = tid >> 5;
  const int b0   = blockIdx.x * (MWAVES * 16) + wave * 16;

  h16* act1 = &s_act[wave][0];        // 16x512
  h16* act2 = act1 + 16 * 512;        // 16x256
  h16* act3 = act1;                   // 16x128 overlays dead act1
  h16* act4 = act2;                   // 16x64  overlays dead act2

  const h16* Wh1 = Whq;
  const h16* Wh2 = Whq + N1SZ;
  const h16* Wh3 = Whq + N1SZ + N2SZ;
  const h16* Wh4 = Whq + N1SZ + N2SZ + N3SZ;
  const h16* inG = trajG + (size_t)b0 * (RNN_T * RNN_H);

  mlp_layer<960, 512, true, true>(inG,  nullptr, Wh1, b1, act1, lane);
  mlp_layer<512, 256, true, true>(act1, nullptr, Wh2, b2, act2, lane);
  mlp_layer<256, 128, true, true>(act2, nullptr, Wh3, b3, act3, lane);
  mlp_layer<128, 64,  true, true>(act3, nullptr, Wh4, b4, act4, lane);

  if (lane < 16) {
    float s = b5[0];
#pragma unroll 1
    for (int k = 0; k < 64; ++k)
      s += (float)act4[lane * 64 + k] * W5[k];
    out[b0 + lane] = s;
  }
}

// ============================================================================
// FUSED fallback (trajectory kept in LDS), templated on pre-converted weights.
// ============================================================================
template <bool PRE>
__global__ __launch_bounds__(32 * FWAVES) void rnn_mlp_kernel(
    const float* __restrict__ x, const float* __restrict__ h0,
    const float* __restrict__ Wih0, const float* __restrict__ Wih,
    const float* __restrict__ Whh, const float* __restrict__ bih,
    const float* __restrict__ bhh,
    const float* __restrict__ W1, const float* __restrict__ b1,
    const float* __restrict__ W2, const float* __restrict__ b2,
    const float* __restrict__ W3, const float* __restrict__ b3,
    const float* __restrict__ W4, const float* __restrict__ b4,
    const float* __restrict__ W5, const float* __restrict__ b5,
    const h16* __restrict__ Whq, float* __restrict__ out, int Bsz) {
  __shared__ __align__(32) h16   s_whh[RNN_L][RNN_H][RNN_H];
  __shared__ __align__(32) h16   s_wih[RNN_L - 1][RNN_H][RNN_H];
  __shared__ __align__(32) h16   s_wih0[RNN_H][RNN_H];
  __shared__ float s_bias[RNN_L][RNN_H];
  __shared__ __align__(32) h16   s_wave[FWAVES][WAVE_HALVES];

  const int tid   = threadIdx.x;
  const int lane  = tid & 31;
  const int wave  = tid >> 5;
  const int b0    = blockIdx.x * (FWAVES * 16) + wave * 16;
  const int mrow  = lane & 15;
  const int mbase = (lane >= 16) ? 8 : 0;

  for (int i = tid; i < RNN_L * RNN_H * RNN_H; i += blockDim.x)
    (&s_whh[0][0][0])[i] = (h16)Whh[i];
  for (int i = tid; i < (RNN_L - 1) * RNN_H * RNN_H; i += blockDim.x)
    (&s_wih[0][0][0])[i] = (h16)Wih[i];
  for (int i = tid; i < RNN_H * RNN_H; i += blockDim.x) {
    int n = i >> 5, k = i & 31;
    (&s_wih0[0][0])[i] = (k < RNN_I) ? (h16)Wih0[n * RNN_I + k] : (h16)0.0f;
  }
  for (int i = tid; i < RNN_L * RNN_H; i += blockDim.x)
    (&s_bias[0][0])[i] = bih[i] + bhh[i];
  __syncthreads();

  h16* traj = &s_wave[wave][0];
  h16* act1 = traj + TRAJ_HALVES;
  h16* hx   = act1 + ACT1_HALVES;

  v16h hfrag[RNN_L];
#pragma unroll
  for (int l = 0; l < RNN_L; ++l)
    hfrag[l] = afrag_gf32(h0 + (size_t)l * Bsz * RNN_H, RNN_H, b0 + mrow, lane);

#pragma unroll 1
  for (int t = 0; t < RNN_T; ++t) {
    v16h xa = {};
    if (lane < 16) {
      const float* xp = x + ((size_t)(b0 + lane) * RNN_T + t) * RNN_I;
      xa[0] = (h16)xp[0]; xa[1] = (h16)xp[1]; xa[2] = (h16)xp[2];
    }
    {
      float hv[2][8];
#pragma unroll
      for (int nt = 0; nt < 2; ++nt) {
        v8f acc = {};
        acc = wmma_f16(xa, bfrag_h16(&s_wih0[0][0], RNN_H, nt * 16 + mrow, 0, lane), acc);
        acc = wmma_f16(hfrag[0], bfrag_h16(&s_whh[0][0][0], RNN_H, nt * 16 + mrow, 0, lane), acc);
        float bv = s_bias[0][nt * 16 + mrow];
#pragma unroll
        for (int r = 0; r < 8; ++r) hv[nt][r] = fast_tanh(acc[r] + bv);
      }
#pragma unroll
      for (int nt = 0; nt < 2; ++nt)
#pragma unroll
        for (int r = 0; r < 8; ++r)
          hx[(mbase + r) * RNN_H + nt * 16 + mrow] = (h16)hv[nt][r];
      __syncthreads();
      hfrag[0] = afrag_h16(hx, RNN_H, 0, lane);
    }
#pragma unroll
    for (int l = 1; l < RNN_L; ++l) {
      float hv[2][8];
#pragma unroll
      for (int nt = 0; nt < 2; ++nt) {
        v8f acc = {};
        acc = wmma_f16(hfrag[l - 1], bfrag_h16(&s_wih[l - 1][0][0], RNN_H, nt * 16 + mrow, 0, lane), acc);
        acc = wmma_f16(hfrag[l],     bfrag_h16(&s_whh[l][0][0],     RNN_H, nt * 16 + mrow, 0, lane), acc);
        float bv = s_bias[l][nt * 16 + mrow];
#pragma unroll
        for (int r = 0; r < 8; ++r) hv[nt][r] = fast_tanh(acc[r] + bv);
      }
      h16* dst    = (l == RNN_L - 1) ? (traj + t * RNN_H) : hx;
      int  dstrid = (l == RNN_L - 1) ? (RNN_T * RNN_H) : RNN_H;
#pragma unroll
      for (int nt = 0; nt < 2; ++nt)
#pragma unroll
        for (int r = 0; r < 8; ++r)
          dst[(mbase + r) * dstrid + nt * 16 + mrow] = (h16)hv[nt][r];
      __syncthreads();
      hfrag[l] = afrag_h16(dst, dstrid, 0, lane);
    }
  }
  __syncthreads();

  const h16* Wh1 = Whq;
  const h16* Wh2 = Whq + N1SZ;
  const h16* Wh3 = Whq + N1SZ + N2SZ;
  const h16* Wh4 = Whq + N1SZ + N2SZ + N3SZ;
  h16* act2 = traj;
  h16* act3 = traj + 8192;
  h16* act4 = act1;
  mlp_layer<960, 512, true, PRE>(traj, W1, Wh1, b1, act1, lane);
  mlp_layer<512, 256, true, PRE>(act1, W2, Wh2, b2, act2, lane);
  mlp_layer<256, 128, true, PRE>(act2, W3, Wh3, b3, act3, lane);
  mlp_layer<128, 64,  true, PRE>(act3, W4, Wh4, b4, act4, lane);

  if (lane < 16) {
    float s = b5[0];
#pragma unroll 1
    for (int k = 0; k < 64; ++k)
      s += (float)act4[lane * 64 + k] * W5[k];
    out[b0 + lane] = s;
  }
}

extern "C" void kernel_launch(void* const* d_in, const int* in_sizes, int n_in,
                              void* d_out, int out_size, void* d_ws, size_t ws_size,
                              hipStream_t stream) {
  const float* x    = (const float*)d_in[0];
  const float* h0   = (const float*)d_in[1];
  const float* Wih0 = (const float*)d_in[2];
  const float* Wih  = (const float*)d_in[3];
  const float* Whh  = (const float*)d_in[4];
  const float* bih  = (const float*)d_in[5];
  const float* bhh  = (const float*)d_in[6];
  const float* W1 = (const float*)d_in[7];  const float* b1 = (const float*)d_in[8];
  const float* W2 = (const float*)d_in[9];  const float* b2 = (const float*)d_in[10];
  const float* W3 = (const float*)d_in[11]; const float* b3 = (const float*)d_in[12];
  const float* W4 = (const float*)d_in[13]; const float* b4 = (const float*)d_in[14];
  const float* W5 = (const float*)d_in[15]; const float* b5 = (const float*)d_in[16];

  int Bsz = in_sizes[0] / (RNN_T * RNN_I);

  const size_t need_w    = (size_t)WTOT * sizeof(h16);               // 1.27 MB
  const size_t need_traj = (size_t)Bsz * RNN_T * RNN_H * sizeof(h16); // ~120 MB
  const int conv_blocks  = (WTOT + 255) / 256;

  if (d_ws != nullptr && ws_size >= need_w + need_traj) {
    // split path: high-occupancy RNN + pure-GEMM MLP head through global/L2
    h16* Whq   = (h16*)d_ws;
    h16* trajG = (h16*)((char*)d_ws + need_w);
    conv_weights_kernel<<<conv_blocks, 256, 0, stream>>>(W1, W2, W3, W4, Whq);
    rnn_traj_kernel<<<Bsz / (RWAVES * 16), 32 * RWAVES, 0, stream>>>(
        x, h0, Wih0, Wih, Whh, bih, bhh, trajG, Bsz);
    mlp_head_kernel<<<Bsz / (MWAVES * 16), 32 * MWAVES, 0, stream>>>(
        trajG, Whq, b1, b2, b3, b4, W5, b5, (float*)d_out);
  } else if (d_ws != nullptr && ws_size >= need_w) {
    h16* Whq = (h16*)d_ws;
    conv_weights_kernel<<<conv_blocks, 256, 0, stream>>>(W1, W2, W3, W4, Whq);
    rnn_mlp_kernel<true><<<Bsz / (FWAVES * 16), 32 * FWAVES, 0, stream>>>(
        x, h0, Wih0, Wih, Whh, bih, bhh, W1, b1, W2, b2, W3, b3, W4, b4, W5, b5,
        Whq, (float*)d_out, Bsz);
  } else {
    rnn_mlp_kernel<false><<<Bsz / (FWAVES * 16), 32 * FWAVES, 0, stream>>>(
        x, h0, Wih0, Wih, Whh, bih, bhh, W1, b1, W2, b2, W3, b3, W4, b4, W5, b5,
        (const h16*)nullptr, (float*)d_out, Bsz);
  }
}
